// NATransformerLayer_52432960749588
// MI455X (gfx1250) — compile-verified, matrix operand order checked by reading, not run on hardware
//
#include <hip/hip_runtime.h>
#include <hip/hip_bf16.h>
#include <math.h>

typedef __attribute__((ext_vector_type(16))) _Float16 v16h;
typedef __attribute__((ext_vector_type(8)))  _Float16 v8h;
typedef __attribute__((ext_vector_type(8)))  float    v8f;
typedef unsigned int su4 __attribute__((ext_vector_type(4)));
typedef unsigned int su8 __attribute__((ext_vector_type(8)));

#define NTOK 12544   // 16*28*28
#define DIM  256
#define HEADS 8
#define HD   32
#define MLP  1024
#define KW   7

// ---------------------------------------------------------------------------
// fp32 -> fp16 weight conversion
// ---------------------------------------------------------------------------
__global__ __launch_bounds__(256)
void nat_cvt_f16(const float* __restrict__ src, _Float16* __restrict__ dst, int n) {
    int i = blockIdx.x * 256 + threadIdx.x;
    if (i < n) dst[i] = (_Float16)src[i];
}

// ---------------------------------------------------------------------------
// LayerNorm over last dim (256), one wave32 per token, 8 channels per lane.
// ---------------------------------------------------------------------------
__global__ __launch_bounds__(256)
void nat_layernorm_f16(const float* __restrict__ x, const float* __restrict__ w,
                       const float* __restrict__ b, _Float16* __restrict__ out,
                       int ntok) {
    int wave = (blockIdx.x << 3) + (threadIdx.x >> 5);
    if (wave >= ntok) return;
    int lane = threadIdx.x & 31;
    const float* xp = x + (size_t)wave * DIM + lane * 8;
    float v[8];
    float s = 0.f;
    #pragma unroll
    for (int i = 0; i < 8; ++i) { v[i] = xp[i]; s += v[i]; }
    #pragma unroll
    for (int off = 16; off > 0; off >>= 1) s += __shfl_xor(s, off, 32);
    float mean = s * (1.f / 256.f);
    float s2 = 0.f;
    #pragma unroll
    for (int i = 0; i < 8; ++i) { float d = v[i] - mean; s2 += d * d; }
    #pragma unroll
    for (int off = 16; off > 0; off >>= 1) s2 += __shfl_xor(s2, off, 32);
    float rstd = rsqrtf(s2 * (1.f / 256.f) + 1e-5f);
    _Float16* op = out + (size_t)wave * DIM + lane * 8;
    const float* wp = w + lane * 8;
    const float* bp = b + lane * 8;
    #pragma unroll
    for (int i = 0; i < 8; ++i)
        op[i] = (_Float16)((v[i] - mean) * rstd * wp[i] + bp[i]);
}

// ---------------------------------------------------------------------------
// WMMA GEMM with Tensor-Data-Mover staging and double-buffered LDS.
//   out[M,N] = A[M,K] (f16 row-major) @ W[N,K]^T (f16) + bias [+ epilogue]
// Block tile 128x128, 8 waves (2 M x 4 N), each wave 64x32 = 4x2 fragments
// of v_wmma_f32_16x16x32_f16.  K staged 32-wide: one TDM per tile per slab,
// LDS row stride 40 f16 produced by TDM padding (16 DW interval, 4 DW pad).
// ---------------------------------------------------------------------------
#define BM 128
#define BN 128
#define BK 32
#define ASTR 40   // f16 per LDS row = 32 data + 8 pad (16B aligned rows)

__device__ __forceinline__ void nat_tdm_issue(unsigned lds_off,
                                              const _Float16* gptr,
                                              int K, int rows) {
    unsigned long long ga = (unsigned long long)(uintptr_t)gptr;
    su4 g0;
    g0[0] = 1u;                                   // count=1, no gather
    g0[1] = lds_off;                              // lds_addr
    g0[2] = (unsigned)ga;                         // global_addr[31:0]
    g0[3] = (unsigned)(ga >> 32) | 0x80000000u;   // global_addr[56:32] | type=2
    su8 g1;
    g1[0] = (1u << 16)      // data_size = 2 bytes
          | (1u << 20)      // pad_enable
          | (3u << 22)      // pad_interval: every 16 DWORDs (64B = one tile row)
          | (3u << 25);     // pad_amount: 4 DWORDs (8 f16) -> row stride 40 f16
    g1[1] = ((unsigned)K & 0xFFFFu) << 16;                         // tensor_dim0 lo
    g1[2] = ((unsigned)K >> 16) | (((unsigned)rows & 0xFFFFu) << 16); // dim0 hi | dim1 lo
    g1[3] = ((unsigned)rows >> 16) | (32u << 16);                  // dim1 hi | tile_dim0=32
    g1[4] = 128u;                                                  // tile_dim1=128
    g1[5] = (unsigned)K;                                           // tensor_dim0_stride lo
    g1[6] = 0u;
    g1[7] = 0u;
    asm volatile("tensor_load_to_lds %0, %1" :: "s"(g0), "s"(g1) : "memory");
}

__device__ __forceinline__ v16h nat_ld_frag(const _Float16* p) {
    v8h lo = *(const v8h*)p;
    v8h hi = *(const v8h*)(p + 16);
    return __builtin_shufflevector(lo, hi, 0,1,2,3,4,5,6,7,8,9,10,11,12,13,14,15);
}

template <int EPI>   // 1: f32 +bias+resid   2: f16 gelu(bias+acc)   3: f16 +bias
__global__ __launch_bounds__(256)
void nat_gemm_wmma(const _Float16* __restrict__ A, const _Float16* __restrict__ W,
                   const float* __restrict__ bias, const float* __restrict__ resid,
                   float* __restrict__ outF, _Float16* __restrict__ outH,
                   int M, int N, int K) {
    __shared__ _Float16 As[2][BM * ASTR];
    __shared__ _Float16 Bs[2][BN * ASTR];

    const int m0   = blockIdx.x * BM;
    const int n0   = blockIdx.y * BN;
    const int tid  = threadIdx.x;
    const int wave = tid >> 5;
    const int lane = tid & 31;
    const int wm   = wave & 1;        // 0..1 (M)
    const int wn   = wave >> 1;       // 0..3 (N)
    const int fm   = lane & 15;
    const int kh   = lane >> 4;       // K-half select

    v8f acc[4][2] = {};

    const _Float16* gA = A + (size_t)m0 * K;
    const _Float16* gB = W + (size_t)n0 * K;

    if (wave == 0) {   // one wave drives the DMA engine
        nat_tdm_issue((unsigned)(uintptr_t)&As[0][0], gA, K, M);
        nat_tdm_issue((unsigned)(uintptr_t)&Bs[0][0], gB, K, N);
    }

    int buf = 0;
    for (int k0 = 0; k0 < K; k0 += BK) {
        __builtin_amdgcn_s_wait_tensorcnt(0);   // issuing wave: DMA landed
        __syncthreads();                        // publish LDS to all waves
        if (wave == 0 && k0 + BK < K) {         // prefetch next slab via TDM
            nat_tdm_issue((unsigned)(uintptr_t)&As[buf ^ 1][0], gA + k0 + BK, K, M);
            nat_tdm_issue((unsigned)(uintptr_t)&Bs[buf ^ 1][0], gB + k0 + BK, K, N);
        }

        v16h bf[2];
        #pragma unroll
        for (int c = 0; c < 2; ++c)
            bf[c] = nat_ld_frag(&Bs[buf][(wn * 32 + c * 16 + fm) * ASTR + kh * 8]);

        #pragma unroll
        for (int r = 0; r < 4; ++r) {
            v16h af = nat_ld_frag(&As[buf][(wm * 64 + r * 16 + fm) * ASTR + kh * 8]);
            #pragma unroll
            for (int c = 0; c < 2; ++c)
                acc[r][c] = __builtin_amdgcn_wmma_f32_16x16x32_f16(
                    false, af, false, bf[c], (short)0, acc[r][c], false, false);
        }
        buf ^= 1;
    }

    // epilogue
    #pragma unroll
    for (int r = 0; r < 4; ++r) {
        const int rowBase = m0 + wm * 64 + r * 16 + kh * 8;
        #pragma unroll
        for (int c = 0; c < 2; ++c) {
            const int col = n0 + wn * 32 + c * 16 + fm;
            const float bv = bias[col];
            #pragma unroll
            for (int vg = 0; vg < 8; ++vg) {
                const int row = rowBase + vg;
                const size_t o = (size_t)row * N + col;
                float v = acc[r][c][vg] + bv;
                if (EPI == 1) {
                    outF[o] = v + resid[o];
                } else if (EPI == 2) {
                    outH[o] = (_Float16)(0.5f * v * (1.f + erff(v * 0.70710678118654752f)));
                } else {
                    outH[o] = (_Float16)v;
                }
            }
        }
    }
}

// ---------------------------------------------------------------------------
// Neighborhood attention, one wave32 per (token, head), lane = head-dim.
// Streaming softmax over the 7x7 clamped window with relative position bias.
// ---------------------------------------------------------------------------
__global__ __launch_bounds__(256)
void nat_attention(const _Float16* __restrict__ qkv, const float* __restrict__ rpb,
                   _Float16* __restrict__ out) {
    const int wave = (blockIdx.x << 3) + (threadIdx.x >> 5);
    const int lane = threadIdx.x & 31;
    const int head = wave & 7;
    const int tok  = wave >> 3;
    const int j = tok % 28;
    const int i = (tok / 28) % 28;
    const int base_bi = tok - i * 28 - j;   // b * 784

    int sh = i - (KW / 2); sh = sh < 0 ? 0 : (sh > 21 ? 21 : sh);
    int sw = j - (KW / 2); sw = sw < 0 ? 0 : (sw > 21 ? 21 : sw);

    const float q = (float)qkv[(size_t)tok * 768 + head * HD + lane] * 0.17677669529663687f;
    const float* rp = rpb + head * 169 + (sh - i + 6) * 13 + (sw - j + 6);

    float m = -1e30f, s = 0.f, acc = 0.f;
    for (int a = 0; a < KW; ++a) {
        const int rowOff = base_bi + (sh + a) * 28;
        for (int c = 0; c < KW; ++c) {
            const int t = rowOff + sw + c;
            const _Float16* kv = qkv + (size_t)t * 768 + head * HD + lane;
            float p = q * (float)kv[DIM];
            #pragma unroll
            for (int off = 16; off > 0; off >>= 1) p += __shfl_xor(p, off, 32);
            const float logit = p + rp[a * 13 + c];
            const float mn = fmaxf(m, logit);
            const float sc = expf(m - mn);
            const float e  = expf(logit - mn);
            s   = s * sc + e;
            acc = acc * sc + e * (float)kv[2 * DIM];
            m = mn;
        }
    }
    out[(size_t)tok * DIM + head * HD + lane] = (_Float16)(acc / s);
}

// ---------------------------------------------------------------------------
// Host launcher
// ---------------------------------------------------------------------------
extern "C" void kernel_launch(void* const* d_in, const int* in_sizes, int n_in,
                              void* d_out, int out_size, void* d_ws, size_t ws_size,
                              hipStream_t stream) {
    const float* x      = (const float*)d_in[0];
    const float* n1w    = (const float*)d_in[1];
    const float* n1b    = (const float*)d_in[2];
    const float* qkv_w  = (const float*)d_in[3];
    const float* qkv_b  = (const float*)d_in[4];
    const float* rpb    = (const float*)d_in[5];
    const float* proj_w = (const float*)d_in[6];
    const float* proj_b = (const float*)d_in[7];
    const float* n2w    = (const float*)d_in[8];
    const float* n2b    = (const float*)d_in[9];
    const float* fc1_w  = (const float*)d_in[10];
    const float* fc1_b  = (const float*)d_in[11];
    const float* fc2_w  = (const float*)d_in[12];
    const float* fc2_b  = (const float*)d_in[13];

    char* ws = (char*)d_ws;
    size_t off = 0;
    auto carve = [&](size_t bytes) -> char* {
        off = (off + 255) & ~(size_t)255;
        char* p = ws + off;
        off += bytes;
        return p;
    };
    _Float16* xn    = (_Float16*)carve((size_t)NTOK * DIM * 2);
    _Float16* qkv16 = (_Float16*)carve((size_t)NTOK * 3 * DIM * 2);
    _Float16* aout  = (_Float16*)carve((size_t)NTOK * DIM * 2);
    float*    xattn = (float*)   carve((size_t)NTOK * DIM * 4);
    _Float16* ln2   = (_Float16*)carve((size_t)NTOK * DIM * 2);
    _Float16* h16   = (_Float16*)carve((size_t)NTOK * MLP * 2);
    _Float16* wq16  = (_Float16*)carve((size_t)3 * DIM * DIM * 2);
    _Float16* wp16  = (_Float16*)carve((size_t)DIM * DIM * 2);
    _Float16* w116  = (_Float16*)carve((size_t)MLP * DIM * 2);
    _Float16* w216  = (_Float16*)carve((size_t)DIM * MLP * 2);

    // 1) weights -> f16
    nat_cvt_f16<<<(3 * DIM * DIM + 255) / 256, 256, 0, stream>>>(qkv_w, wq16, 3 * DIM * DIM);
    nat_cvt_f16<<<(DIM * DIM + 255) / 256, 256, 0, stream>>>(proj_w, wp16, DIM * DIM);
    nat_cvt_f16<<<(MLP * DIM + 255) / 256, 256, 0, stream>>>(fc1_w, w116, MLP * DIM);
    nat_cvt_f16<<<(DIM * MLP + 255) / 256, 256, 0, stream>>>(fc2_w, w216, DIM * MLP);

    // 2) LN1 -> f16
    nat_layernorm_f16<<<NTOK / 8, 256, 0, stream>>>(x, n1w, n1b, xn, NTOK);

    // 3) QKV GEMM: [12544,256] x [768,256]^T -> f16 [12544,768]
    nat_gemm_wmma<3><<<dim3(NTOK / BM, (3 * DIM) / BN), 256, 0, stream>>>(
        xn, wq16, qkv_b, nullptr, nullptr, qkv16, NTOK, 3 * DIM, DIM);

    // 4) neighborhood attention
    nat_attention<<<NTOK, 256, 0, stream>>>(qkv16, rpb, aout);

    // 5) proj GEMM + residual(x) -> f32 xattn
    nat_gemm_wmma<1><<<dim3(NTOK / BM, DIM / BN), 256, 0, stream>>>(
        aout, wp16, proj_b, x, xattn, nullptr, NTOK, DIM, DIM);

    // 6) LN2 -> f16
    nat_layernorm_f16<<<NTOK / 8, 256, 0, stream>>>(xattn, n2w, n2b, ln2, NTOK);

    // 7) FC1 GEMM + exact GELU -> f16 [12544,1024]
    nat_gemm_wmma<2><<<dim3(NTOK / BM, MLP / BN), 256, 0, stream>>>(
        ln2, w116, fc1_b, nullptr, nullptr, h16, NTOK, MLP, DIM);

    // 8) FC2 GEMM + residual(xattn) -> f32 d_out
    nat_gemm_wmma<1><<<dim3(NTOK / BM, DIM / BN), 256, 0, stream>>>(
        h16, w216, fc2_b, xattn, (float*)d_out, nullptr, NTOK, DIM, MLP);
}
